// LexJadeAttention_16157666968010
// MI455X (gfx1250) — compile-verified, matrix operand order checked by reading, not run on hardware
//
#include <hip/hip_runtime.h>

typedef __bf16 bf16;
typedef __attribute__((ext_vector_type(16))) __bf16 v16bf;
typedef __attribute__((ext_vector_type(8)))  __bf16 v8bf;
typedef __attribute__((ext_vector_type(4)))  __bf16 v4bf;
typedef __attribute__((ext_vector_type(8)))  float  v8f;

#define BB  2
#define SS  2048
#define HH  2048
#define NH  16
#define NKV 4
#define HD  128

static __device__ inline v16bf ld_frag(const bf16* p) {
  // elements 0..7 <- p[0..7], elements 8..15 <- p[16..23]
  // (caller adds koff = 0 or 8 per lane half, per 16-bit A/B VGPR layout)
  v8bf lo = *(const v8bf*)(p);
  v8bf hi = *(const v8bf*)(p + 16);
  v16bf r;
#pragma unroll
  for (int i = 0; i < 8; ++i) { r[i] = lo[i]; r[i + 8] = hi[i]; }
  return r;
}

static __device__ inline v8f wmma_bf16(v16bf a, v16bf b, v8f c) {
  return __builtin_amdgcn_wmma_f32_16x16x32_bf16(false, a, false, b, (short)0, c,
                                                 false, false);
}

static __device__ inline v8f vzero8() {
  v8f z;
#pragma unroll
  for (int i = 0; i < 8; ++i) z[i] = 0.0f;
  return z;
}

static __device__ inline float rmax16(float x) {
  x = fmaxf(x, __shfl_xor(x, 1));
  x = fmaxf(x, __shfl_xor(x, 2));
  x = fmaxf(x, __shfl_xor(x, 4));
  x = fmaxf(x, __shfl_xor(x, 8));
  return x;
}

static __device__ inline float rsum16(float x) {
  x += __shfl_xor(x, 1);
  x += __shfl_xor(x, 2);
  x += __shfl_xor(x, 4);
  x += __shfl_xor(x, 8);
  return x;
}

// Low 32 bits of a generic pointer into __shared__ are the LDS byte offset
// (flat LDS aperture truncates to addr[31:0], ISA 10.2).
static __device__ inline unsigned ldsOff(const void* p) {
  return (unsigned)(size_t)p;
}

// Per-lane async 16B copy global -> LDS (ASYNCcnt path, gfx1250).
static __device__ inline void async_ld16(unsigned lds, const bf16* g) {
  asm volatile("global_load_async_to_lds_b128 %0, %1, off"
               :: "v"(lds), "v"(g)
               : "memory");
}
static __device__ inline void wait_async0() {
  asm volatile("s_wait_asynccnt 0" ::: "memory");
}

// ---------------- fp32 -> bf16 convert (vectorized) ----------------
__global__ void k_conv_x(const float* __restrict__ x, bf16* __restrict__ y, int n4) {
  int i = blockIdx.x * blockDim.x + threadIdx.x;
  if (i >= n4) return;
  float4 f = ((const float4*)x)[i];
  v4bf r;
  r[0] = (bf16)f.x; r[1] = (bf16)f.y; r[2] = (bf16)f.z; r[3] = (bf16)f.w;
  *(v4bf*)(y + 4 * (size_t)i) = r;
}

// ---------------- W[K][N] fp32 -> Wt[N][K] bf16 (LDS tiled) ----------------
__global__ void k_wtrans(const float* __restrict__ w, bf16* __restrict__ wt,
                         int K, int N) {
  __shared__ float tile[64][65];
  const int bx = blockIdx.x * 64;          // n
  const int by = blockIdx.y * 64;          // k
  const int c  = threadIdx.x & 63;
  const int rb = (threadIdx.x >> 6) * 16;
#pragma unroll
  for (int i = 0; i < 16; ++i)
    tile[rb + i][c] = w[(size_t)(by + rb + i) * N + bx + c];
  __syncthreads();
#pragma unroll
  for (int i = 0; i < 16; ++i)
    wt[(size_t)(bx + rb + i) * K + by + c] = (bf16)tile[c][rb + i];
}

// ---------------- QKV projection GEMM + fused RoPE ----------------
// X:[B*S][K] bf16, Wt:[nHeads*128][K] bf16, Out:[(b*nHeads+h)*S + s]*128 bf16
__global__ __launch_bounds__(256) void k_gemm_qkv(
    const bf16* __restrict__ X, const bf16* __restrict__ Wt,
    const float* __restrict__ cs, const float* __restrict__ sn,
    bf16* __restrict__ Out, int Kdim, int nHeads, int applyRope) {
  __shared__ __align__(16) bf16 As[128][40];   // pad -> conflict-free b128 frag reads
  __shared__ __align__(16) bf16 Bs[128][40];
  __shared__ __align__(16) bf16 Cs[128][128];

  const int tid  = threadIdx.x;
  const int lane = tid & 31;
  const int w    = tid >> 5;
  const int wm   = w & 3;                  // 4 waves along M (32 rows each)
  const int wn   = w >> 2;                 // 2 waves along N (64 cols each)
  const int m0   = blockIdx.x * 128;
  const int h    = blockIdx.y;
  const bf16* Wh = Wt + (size_t)h * 128 * Kdim;

  const int r    = lane & 15;
  const int hi   = lane >> 4;
  const int koff = hi ? 8 : 0;

  v8f acc[2][4];
#pragma unroll
  for (int a = 0; a < 2; ++a)
#pragma unroll
    for (int q = 0; q < 4; ++q) acc[a][q] = vzero8();

  const int stR = tid >> 1;
  const int stC = (tid & 1) * 16;

  for (int k0 = 0; k0 < Kdim; k0 += 32) {
    __syncthreads();
    {
      const bf16* srcA = X + (size_t)(m0 + stR) * Kdim + k0 + stC;
      *(v8bf*)&As[stR][stC]     = *(const v8bf*)(srcA);
      *(v8bf*)&As[stR][stC + 8] = *(const v8bf*)(srcA + 8);
      const bf16* srcB = Wh + (size_t)stR * Kdim + k0 + stC;
      *(v8bf*)&Bs[stR][stC]     = *(const v8bf*)(srcB);
      *(v8bf*)&Bs[stR][stC + 8] = *(const v8bf*)(srcB + 8);
    }
    __syncthreads();

    v16bf a0 = ld_frag(&As[32 * wm + r][koff]);
    v16bf a1 = ld_frag(&As[32 * wm + 16 + r][koff]);
    v16bf b0 = ld_frag(&Bs[64 * wn + r][koff]);
    v16bf b1 = ld_frag(&Bs[64 * wn + 16 + r][koff]);
    v16bf b2 = ld_frag(&Bs[64 * wn + 32 + r][koff]);
    v16bf b3 = ld_frag(&Bs[64 * wn + 48 + r][koff]);

    acc[0][0] = wmma_bf16(a0, b0, acc[0][0]);
    acc[0][1] = wmma_bf16(a0, b1, acc[0][1]);
    acc[0][2] = wmma_bf16(a0, b2, acc[0][2]);
    acc[0][3] = wmma_bf16(a0, b3, acc[0][3]);
    acc[1][0] = wmma_bf16(a1, b0, acc[1][0]);
    acc[1][1] = wmma_bf16(a1, b1, acc[1][1]);
    acc[1][2] = wmma_bf16(a1, b2, acc[1][2]);
    acc[1][3] = wmma_bf16(a1, b3, acc[1][3]);
  }

  __syncthreads();
#pragma unroll
  for (int mt = 0; mt < 2; ++mt)
#pragma unroll
    for (int nt = 0; nt < 4; ++nt)
#pragma unroll
      for (int v = 0; v < 8; ++v)
        Cs[32 * wm + 16 * mt + 8 * hi + v][64 * wn + 16 * nt + r] =
            (bf16)acc[mt][nt][v];
  __syncthreads();

  const int row = tid >> 1;
  const int cb  = (tid & 1) * 64;
  const int sg  = m0 + row;
  const int b   = sg / SS;
  const int s   = sg % SS;
  bf16* outp = Out + ((size_t)(b * nHeads + h) * SS + s) * HD;
  if (applyRope) {
#pragma unroll 8
    for (int j0 = 0; j0 < 64; ++j0) {
      const int j = cb + j0;
      float vv  = (float)Cs[row][j];
      float rot = (j < 64) ? -(float)Cs[row][j + 64] : (float)Cs[row][j - 64];
      outp[j] = (bf16)(vv * cs[s * HD + j] + rot * sn[s * HD + j]);
    }
  } else {
#pragma unroll 8
    for (int j0 = 0; j0 < 64; ++j0) {
      const int j = cb + j0;
      outp[j] = Cs[row][j];
    }
  }
}

// ---------------- sliding-window flash attention (all WMMA) ----------------
// Q:[b,16,S,128], K/V:[b,4,S,128] bf16; Oa:[B*S][2048] bf16
__global__ __launch_bounds__(256) void k_attn(
    const bf16* __restrict__ Q, const bf16* __restrict__ Km,
    const bf16* __restrict__ V, bf16* __restrict__ Oa,
    const int* __restrict__ swp) {
  __shared__ __align__(16) bf16 Ks[64][136];   // K rows (key-major), padded
  __shared__ __align__(16) bf16 Vs[128][72];   // V transposed (hd-major), padded
  __shared__ __align__(16) bf16 Ps[8][16][72]; // per-wave P tile for C->A relayout

  const int window = swp[0];
  const int qb = blockIdx.x, h = blockIdx.y, b = blockIdx.z;
  const int hk = h >> 2;                       // repeat_interleave GQA map
  const int q0 = qb * 128;
  const int tid = threadIdx.x, lane = tid & 31, w = tid >> 5;
  const int r = lane & 15, hi = lane >> 4, koff = hi ? 8 : 0;

  const bf16* Qb_ = Q  + ((size_t)(b * NH  + h)  * SS + q0) * HD;
  const bf16* Kb_ = Km + ((size_t)(b * NKV + hk) * SS) * HD;
  const bf16* Vb_ = V  + ((size_t)(b * NKV + hk) * SS) * HD;

  // Q A-fragments: resident for the whole q-block (wave w owns q rows 16w..16w+15)
  v16bf aq[4];
#pragma unroll
  for (int c = 0; c < 4; ++c)
    aq[c] = ld_frag(Qb_ + (size_t)(16 * w + r) * HD + 32 * c + koff);

  v8f o[8];
#pragma unroll
  for (int t = 0; t < 8; ++t) o[t] = vzero8();
  float mrow[8], lrow[8];
#pragma unroll
  for (int v = 0; v < 8; ++v) { mrow[v] = -1.0e30f; lrow[v] = 0.0f; }

  const int kmin = q0 - window + 1;
  const int kbF  = (kmin <= 0) ? 0 : (kmin >> 6);
  const int kbL  = (q0 + 127) >> 6;

  for (int kb = kbF; kb <= kbL; ++kb) {
    const int k0 = kb << 6;
    __syncthreads();
    {
      const int row = tid >> 2, cbk = (tid & 3) * 32;
      // K tile: pure copy -> gfx1250 async global->LDS DMA (ASYNCcnt)
      const bf16* srcK = Kb_ + (size_t)(k0 + row) * HD + cbk;
#pragma unroll
      for (int u = 0; u < 4; ++u)
        async_ld16(ldsOff(&Ks[row][cbk + 8 * u]), srcK + 8 * u);
      // V tile: transposing scatter, regular path
      const bf16* srcV = Vb_ + (size_t)(k0 + row) * HD + cbk;
#pragma unroll
      for (int u = 0; u < 4; ++u) {
        v8bf xv = *(const v8bf*)(srcV + 8 * u);
#pragma unroll
        for (int j = 0; j < 8; ++j) Vs[cbk + 8 * u + j][row] = xv[j];
      }
    }
    if (kb < kbL) {                      // pull next K/V block toward caches
      __builtin_prefetch(Kb_ + (size_t)(k0 + 64) * HD + tid * 32, 0, 1);
      __builtin_prefetch(Vb_ + (size_t)(k0 + 64) * HD + tid * 32, 0, 1);
    }
    wait_async0();                       // K DMA complete before barrier
    __syncthreads();

    // S = Q @ K^T : B-fragment lane = key row of Ks (contiguous hd chunk)
    v8f sc[4];
#pragma unroll
    for (int nt = 0; nt < 4; ++nt) {
      v8f cacc = vzero8();
#pragma unroll
      for (int c = 0; c < 4; ++c) {
        v16bf bk = ld_frag(&Ks[16 * nt + r][32 * c + koff]);
        cacc = wmma_bf16(aq[c], bk, cacc);
      }
      sc[nt] = cacc;
    }

    // scale + sliding-window/causal mask (rel = ki - qi; keep -window < rel <= 0)
#pragma unroll
    for (int nt = 0; nt < 4; ++nt)
#pragma unroll
      for (int v = 0; v < 8; ++v) {
        const int qi = q0 + 16 * w + 8 * hi + v;
        const int ki = k0 + 16 * nt + r;
        const float sv = sc[nt][v] * 0.08838834764831845f;  // 1/sqrt(128)
        sc[nt][v] = (ki <= qi && ki > qi - window) ? sv : -1.0e30f;
      }

    // online softmax stats (16-lane halves hold rows v / 8+v of the C layout)
    float rs[8];
#pragma unroll
    for (int v = 0; v < 8; ++v) {
      float mx = fmaxf(fmaxf(sc[0][v], sc[1][v]), fmaxf(sc[2][v], sc[3][v]));
      mx = rmax16(mx);
      const float mnew = fmaxf(mrow[v], mx);
      rs[v] = __expf(mrow[v] - mnew);
      mrow[v] = mnew;
    }
#pragma unroll
    for (int nt = 0; nt < 4; ++nt)
#pragma unroll
      for (int v = 0; v < 8; ++v)
        sc[nt][v] = __expf(sc[nt][v] - mrow[v]);
#pragma unroll
    for (int v = 0; v < 8; ++v) {
      float sum = sc[0][v] + sc[1][v] + sc[2][v] + sc[3][v];
      sum = rsum16(sum);
      lrow[v] = lrow[v] * rs[v] + sum;
    }
#pragma unroll
    for (int t2 = 0; t2 < 8; ++t2)
#pragma unroll
      for (int v = 0; v < 8; ++v) o[t2][v] *= rs[v];

    // P: C layout -> LDS -> A layout (per-wave buffer, wave-internal ordering)
#pragma unroll
    for (int nt = 0; nt < 4; ++nt)
#pragma unroll
      for (int v = 0; v < 8; ++v)
        Ps[w][8 * hi + v][16 * nt + r] = (bf16)sc[nt][v];

    v16bf ap0 = ld_frag(&Ps[w][r][koff]);
    v16bf ap1 = ld_frag(&Ps[w][r][32 + koff]);
    // O += P @ V : B-fragment lane = hd column of Vs (contiguous key chunk)
#pragma unroll
    for (int t2 = 0; t2 < 8; ++t2) {
      v16bf bv0 = ld_frag(&Vs[16 * t2 + r][koff]);
      o[t2] = wmma_bf16(ap0, bv0, o[t2]);
      v16bf bv1 = ld_frag(&Vs[16 * t2 + r][32 + koff]);
      o[t2] = wmma_bf16(ap1, bv1, o[t2]);
    }
  }

#pragma unroll
  for (int v = 0; v < 8; ++v) lrow[v] = 1.0f / lrow[v];
  const int qrow = q0 + 16 * w + 8 * hi;
#pragma unroll
  for (int t2 = 0; t2 < 8; ++t2)
#pragma unroll
    for (int v = 0; v < 8; ++v) {
      const size_t idx =
          (size_t)(b * SS + qrow + v) * (NH * HD) + h * HD + 16 * t2 + r;
      Oa[idx] = (bf16)(o[t2][v] * lrow[v]);
    }
}

// ---------------- output projection GEMM (fp32 epilogue) ----------------
__global__ __launch_bounds__(256) void k_gemm_out(
    const bf16* __restrict__ X, const bf16* __restrict__ Wt,
    float* __restrict__ Yout, int Kdim) {
  __shared__ __align__(16) bf16 As[128][40];
  __shared__ __align__(16) bf16 Bs[128][40];

  const int tid  = threadIdx.x;
  const int lane = tid & 31;
  const int w    = tid >> 5;
  const int wm   = w & 3;
  const int wn   = w >> 2;
  const int m0   = blockIdx.x * 128;
  const int n0   = blockIdx.y * 128;
  const bf16* Wh = Wt + (size_t)n0 * Kdim;

  const int r    = lane & 15;
  const int hi   = lane >> 4;
  const int koff = hi ? 8 : 0;

  v8f acc[2][4];
#pragma unroll
  for (int a = 0; a < 2; ++a)
#pragma unroll
    for (int q = 0; q < 4; ++q) acc[a][q] = vzero8();

  const int stR = tid >> 1;
  const int stC = (tid & 1) * 16;

  for (int k0 = 0; k0 < Kdim; k0 += 32) {
    __syncthreads();
    {
      const bf16* srcA = X + (size_t)(m0 + stR) * Kdim + k0 + stC;
      *(v8bf*)&As[stR][stC]     = *(const v8bf*)(srcA);
      *(v8bf*)&As[stR][stC + 8] = *(const v8bf*)(srcA + 8);
      const bf16* srcB = Wh + (size_t)stR * Kdim + k0 + stC;
      *(v8bf*)&Bs[stR][stC]     = *(const v8bf*)(srcB);
      *(v8bf*)&Bs[stR][stC + 8] = *(const v8bf*)(srcB + 8);
    }
    __syncthreads();

    v16bf a0 = ld_frag(&As[32 * wm + r][koff]);
    v16bf a1 = ld_frag(&As[32 * wm + 16 + r][koff]);
    v16bf b0 = ld_frag(&Bs[64 * wn + r][koff]);
    v16bf b1 = ld_frag(&Bs[64 * wn + 16 + r][koff]);
    v16bf b2 = ld_frag(&Bs[64 * wn + 32 + r][koff]);
    v16bf b3 = ld_frag(&Bs[64 * wn + 48 + r][koff]);

    acc[0][0] = wmma_bf16(a0, b0, acc[0][0]);
    acc[0][1] = wmma_bf16(a0, b1, acc[0][1]);
    acc[0][2] = wmma_bf16(a0, b2, acc[0][2]);
    acc[0][3] = wmma_bf16(a0, b3, acc[0][3]);
    acc[1][0] = wmma_bf16(a1, b0, acc[1][0]);
    acc[1][1] = wmma_bf16(a1, b1, acc[1][1]);
    acc[1][2] = wmma_bf16(a1, b2, acc[1][2]);
    acc[1][3] = wmma_bf16(a1, b3, acc[1][3]);
  }

#pragma unroll
  for (int mt = 0; mt < 2; ++mt)
#pragma unroll
    for (int nt = 0; nt < 4; ++nt)
#pragma unroll
      for (int v = 0; v < 8; ++v) {
        const size_t row = (size_t)(m0 + 32 * wm + 16 * mt + 8 * hi + v);
        Yout[row * HH + n0 + 64 * wn + 16 * nt + r] = acc[mt][nt][v];
      }
}

// ---------------- host launcher ----------------
extern "C" void kernel_launch(void* const* d_in, const int* in_sizes, int n_in,
                              void* d_out, int out_size, void* d_ws, size_t ws_size,
                              hipStream_t stream) {
  (void)in_sizes; (void)n_in; (void)out_size; (void)ws_size;
  const float* X  = (const float*)d_in[0];
  const float* cs = (const float*)d_in[1];
  const float* sn = (const float*)d_in[2];
  const float* Wq = (const float*)d_in[3];
  const float* Wk = (const float*)d_in[4];
  const float* Wv = (const float*)d_in[5];
  const float* Wo = (const float*)d_in[6];
  const int*   sw = (const int*)d_in[7];

  char* ws = (char*)d_ws;
  size_t off = 0;
  auto alloc = [&](size_t bytes) -> void* {
    void* p = ws + off;
    off = (off + bytes + 255) & ~(size_t)255;
    return p;
  };
  bf16* Xb  = (bf16*)alloc((size_t)BB * SS * HH * 2);
  bf16* Wqt = (bf16*)alloc((size_t)HH * HH * 2);
  bf16* Wkt = (bf16*)alloc((size_t)(NKV * HD) * HH * 2);
  bf16* Wvt = (bf16*)alloc((size_t)(NKV * HD) * HH * 2);
  bf16* Wot = (bf16*)alloc((size_t)HH * HH * 2);
  bf16* Qb  = (bf16*)alloc((size_t)BB * NH  * SS * HD * 2);
  bf16* Kb  = (bf16*)alloc((size_t)BB * NKV * SS * HD * 2);
  bf16* Vb  = (bf16*)alloc((size_t)BB * NKV * SS * HD * 2);
  bf16* Ao  = (bf16*)alloc((size_t)BB * SS * NH * HD * 2);

  const int nx4 = BB * SS * HH / 4;
  k_conv_x<<<(nx4 + 255) / 256, 256, 0, stream>>>(X, Xb, nx4);

  k_wtrans<<<dim3(HH / 64, HH / 64), 256, 0, stream>>>(Wq, Wqt, HH, HH);
  k_wtrans<<<dim3((NKV * HD) / 64, HH / 64), 256, 0, stream>>>(Wk, Wkt, HH, NKV * HD);
  k_wtrans<<<dim3((NKV * HD) / 64, HH / 64), 256, 0, stream>>>(Wv, Wvt, HH, NKV * HD);
  k_wtrans<<<dim3(HH / 64, HH / 64), 256, 0, stream>>>(Wo, Wot, HH, HH);

  k_gemm_qkv<<<dim3(BB * SS / 128, NH), 256, 0, stream>>>(Xb, Wqt, cs, sn, Qb, HH, NH, 1);
  k_gemm_qkv<<<dim3(BB * SS / 128, NKV), 256, 0, stream>>>(Xb, Wkt, cs, sn, Kb, HH, NKV, 1);
  k_gemm_qkv<<<dim3(BB * SS / 128, NKV), 256, 0, stream>>>(Xb, Wvt, cs, sn, Vb, HH, NKV, 0);

  k_attn<<<dim3(SS / 128, NH, BB), 256, 0, stream>>>(Qb, Kb, Vb, Ao, sw);

  k_gemm_out<<<dim3(BB * SS / 128, HH / 128), 256, 0, stream>>>(Ao, Wot, (float*)d_out, HH);
}
